// DeepContextDependentChoice_85521388798585
// MI455X (gfx1250) — matrix operand order repaired and checked by reading, not compiled, not run on hardware
//
#include <hip/hip_runtime.h>

// Shapes from the reference: B=64, A=128, F=32, H=64.
#define B_ 64
#define A_ 128
#define F_ 32
#define H_ 64

typedef __attribute__((ext_vector_type(16))) _Float16 v16h;
typedef __attribute__((ext_vector_type(8)))  _Float16 v8h;
typedef __attribute__((ext_vector_type(8)))  float    v8f;

// ---------------------------------------------------------------------------
// K1: base MLP  base_u[b,a] = W3 . relu(W2 . relu(W1 . x + b1) + b2) + b3
// Tiny (0.1 GFLOP) -> plain VALU, weights staged in LDS (uniform broadcast).
// ---------------------------------------------------------------------------
__global__ void __launch_bounds__(256)
base_kernel(const float* __restrict__ X,  const float* __restrict__ W1,
            const float* __restrict__ b1, const float* __restrict__ W2,
            const float* __restrict__ b2, const float* __restrict__ W3,
            const float* __restrict__ b3, float* __restrict__ base_u)
{
    __shared__ float sW1[F_ * H_];
    __shared__ float sW2[H_ * H_];
    __shared__ float sW3[H_];
    __shared__ float sb1[H_];
    __shared__ float sb2[H_];

    const int tid = threadIdx.x;
    for (int i = tid; i < F_ * H_; i += 256) sW1[i] = W1[i];
    for (int i = tid; i < H_ * H_; i += 256) sW2[i] = W2[i];
    if (tid < H_) { sW3[tid] = W3[tid]; sb1[tid] = b1[tid]; sb2[tid] = b2[tid]; }
    __syncthreads();

    const int gid = blockIdx.x * 256 + tid;       // gid in [0, B*A)
    const float* xr = X + gid * F_;
    float x[F_];
#pragma unroll
    for (int f = 0; f < F_; ++f) x[f] = xr[f];

    float h1[H_];
#pragma unroll 4
    for (int h = 0; h < H_; ++h) {
        float a = sb1[h];
#pragma unroll
        for (int f = 0; f < F_; ++f) a += x[f] * sW1[f * H_ + h];
        h1[h] = fmaxf(a, 0.f);
    }
    float u = b3[0];
#pragma unroll 4
    for (int k = 0; k < H_; ++k) {
        float a = sb2[k];
#pragma unroll
        for (int h = 0; h < H_; ++h) a += h1[h] * sW2[h * H_ + k];
        u += fmaxf(a, 0.f) * sW3[k];
    }
    base_u[gid] = u;
}

// ---------------------------------------------------------------------------
// K2: pairwise MLP via WMMA. Block = (b, 16-row i-tile), 4 waves.
// Each wave owns j = wave (mod 4) and computes ALL 64 hidden columns
// (8 x v_wmma_f32_16x16x32_f16 per j). Masked-out j's are skipped with a
// uniform branch. Layer-2 bias folded into the WMMA C operand; diagonal
// handled only in the in-tile j-range; W3 scale + pair_b3 hoisted out.
// ---------------------------------------------------------------------------
__global__ void __launch_bounds__(128)
pair_kernel(const float* __restrict__ X,   const int*   __restrict__ mask,
            const float* __restrict__ pW1, const float* __restrict__ pb1,
            const float* __restrict__ pW2, const float* __restrict__ pb2,
            const float* __restrict__ pW3, const float* __restrict__ pb3,
            const float* __restrict__ base_u, float* __restrict__ util)
{
    __shared__ _Float16 s_hj[A_ * H_];            // 16 KB  hj = X@Wj (f16)
    __shared__ float    s_hi[16 * H_];            //  4 KB  hi tile (fp32)
    __shared__ __align__(16) char s_ovl[32768];   // 32 KB  overlaid setup/steady
    // setup phase layout:
    float*    s_X    = (float*)s_ovl;                   // A_*F_ floats (16 KB)
    float*    s_PW1  = (float*)(s_ovl + 16384);         // 2F_*H_ floats (16 KB)
    // steady phase layout (written after setup reads complete):
    _Float16* s_W2h  = (_Float16*)s_ovl;                // H_*H_ halves (8 KB)
    int*      s_mask = (int*)(s_ovl + 8192);            // 128 ints
    float*    s_part = (float*)(s_ovl + 8192 + 512);    // 64 floats

    const int b     = blockIdx.y;
    const int i0    = blockIdx.x * 16;
    const int tid   = threadIdx.x;
    const int lane  = tid & 31;
    const int wv    = tid >> 5;        // wave id -> j stratum
    const int khalf = lane >> 4;       // K-half select per ISA 16-bit layouts
    const int m     = lane & 15;       // A row / B,C column within tile

    // --- phase 1: stage X[b] and pair_W1 ---
    const float* Xb = X + b * A_ * F_;
    for (int i = tid; i < A_ * F_;     i += 128) s_X[i]   = Xb[i];
    for (int i = tid; i < 2 * F_ * H_; i += 128) s_PW1[i] = pW1[i];
    __syncthreads();

    // --- phase 2: hj = X@Wj (all 128 rows), hi = X@Wi (this i-tile) ---
    for (int e = 0; e < (A_ * H_) / 128; ++e) {
        int idx = e * 128 + tid;
        int a = idx >> 6, h = idx & 63;
        float acc = 0.f;
#pragma unroll
        for (int f = 0; f < F_; ++f) acc += s_X[a * F_ + f] * s_PW1[(F_ + f) * H_ + h];
        s_hj[idx] = (_Float16)acc;
    }
    for (int e = 0; e < (16 * H_) / 128; ++e) {
        int idx = e * 128 + tid;
        int r = idx >> 6, h = idx & 63;
        float acc = 0.f;
#pragma unroll
        for (int f = 0; f < F_; ++f) acc += s_X[(i0 + r) * F_ + f] * s_PW1[f * H_ + h];
        s_hi[idx] = acc;
    }
    __syncthreads();

    // --- phase 3: overlay now becomes W2(f16) + mask ---
    for (int i = tid; i < H_ * H_; i += 128) s_W2h[i] = (_Float16)pW2[i];
    if (tid < A_) s_mask[tid] = mask[b * A_ + tid];
    __syncthreads();

    // Per-lane loop-invariant state.
    float w3n[4];
    v8f   cb[4];                        // layer-2 bias as WMMA C-init (row splat)
#pragma unroll
    for (int t = 0; t < 4; ++t) {
        w3n[t] = pW3[t * 16 + m];
        float bv = pb2[t * 16 + m];
#pragma unroll
        for (int v = 0; v < 8; ++v) cb[t][v] = bv;
    }

    // hi row + b1, packed f16 in A-operand k-order: group g covers
    // k = g*16 + khalf*8 + [0,8)  (g=0,1 -> A0 elements; g=2,3 -> A1).
    v8h hib[4];
#pragma unroll
    for (int g = 0; g < 4; ++g) {
        int kb = g * 16 + khalf * 8;
#pragma unroll
        for (int t = 0; t < 8; ++t)
            hib[g][t] = (_Float16)(s_hi[m * H_ + kb + t] + pb1[kb + t]);
    }

    // B operands for all four 16-column tiles (table order: elem e -> K=khalf*8+(e&7)+(e>>3)*16).
    v16h bop0[4], bop1[4];
#pragma unroll
    for (int t = 0; t < 4; ++t) {
#pragma unroll
        for (int e = 0; e < 16; ++e) {
            bop0[t][e] = s_W2h[(khalf * 8 + (e & 7) + (e >> 3) * 16) * H_ + t * 16 + m];
            bop1[t][e] = s_W2h[(32 + khalf * 8 + (e & 7) + (e >> 3) * 16) * H_ + t * 16 + m];
        }
    }

    float acc[4][8];
#pragma unroll
    for (int t = 0; t < 4; ++t)
#pragma unroll
        for (int v = 0; v < 8; ++v) acc[t][v] = 0.f;

    const v8h z8 = {};
    // Shared loop body; `diag` is a literal at each call site -> specialized.
    auto body = [&](int j, bool diag) {
        const _Float16* hj = &s_hj[j * H_];
        v8h q0 = *(const v8h*)(hj +      khalf * 8);   // ds_load_b128 each
        v8h q1 = *(const v8h*)(hj + 16 + khalf * 8);
        v8h q2 = *(const v8h*)(hj + 32 + khalf * 8);
        v8h q3 = *(const v8h*)(hj + 48 + khalf * 8);
        v8h h0 = __builtin_elementwise_max(hib[0] + q0, z8);
        v8h h1 = __builtin_elementwise_max(hib[1] + q1, z8);
        v8h h2 = __builtin_elementwise_max(hib[2] + q2, z8);
        v8h h3 = __builtin_elementwise_max(hib[3] + q3, z8);
        v16h a0 = __builtin_shufflevector(h0, h1, 0,1,2,3,4,5,6,7,8,9,10,11,12,13,14,15);
        v16h a1 = __builtin_shufflevector(h2, h3, 0,1,2,3,4,5,6,7,8,9,10,11,12,13,14,15);
#pragma unroll
        for (int t = 0; t < 4; ++t) {
            v8f c = __builtin_amdgcn_wmma_f32_16x16x32_f16(false, a0, false, bop0[t],
                                                           (short)0, cb[t], false, false);
            c = __builtin_amdgcn_wmma_f32_16x16x32_f16(false, a1, false, bop1[t],
                                                       (short)0, c, false, false);
#pragma unroll
            for (int v = 0; v < 8; ++v) {
                float r = fmaxf(c[v], 0.f);
                if (diag && (i0 + v + 8 * khalf == j)) r = 0.f;   // zero pe(i,i)
                acc[t][v] += r;
            }
        }
    };

    // j-ranges: only the in-tile range can hit the diagonal.
#pragma unroll 1
    for (int j = wv; j < i0; j += 4)            { if (s_mask[j]) body(j, false); }
#pragma unroll 1
    for (int j = i0 + wv; j < i0 + 16; j += 4)  { if (s_mask[j]) body(j, true); }
#pragma unroll 1
    for (int j = i0 + 16 + wv; j < A_; j += 4)  { if (s_mask[j]) body(j, false); }

    // Hoisted W3 scale, then in-lane sum over the 4 column tiles.
    float s[8];
#pragma unroll
    for (int v = 0; v < 8; ++v)
        s[v] = acc[0][v] * w3n[0] + acc[1][v] * w3n[1]
             + acc[2][v] * w3n[2] + acc[3][v] * w3n[3];

    // Reduce over the 16 column lanes sharing the same khalf
    // (xor masks 1,2,4,8 never cross the khalf bit = 16).
#pragma unroll
    for (int v = 0; v < 8; ++v) {
        s[v] += __shfl_xor(s[v], 1, 32);
        s[v] += __shfl_xor(s[v], 2, 32);
        s[v] += __shfl_xor(s[v], 4, 32);
        s[v] += __shfl_xor(s[v], 8, 32);
    }
    if (m == 0) {
#pragma unroll
        for (int v = 0; v < 8; ++v) s_part[wv * 16 + khalf * 8 + v] = s[v];
    }
    __syncthreads();

    if (tid < 16) {
        float sum = s_part[tid] + s_part[16 + tid] + s_part[32 + tid] + s_part[48 + tid];
        int nm = 0;
        for (int j = 0; j < A_; ++j) nm += s_mask[j];   // # masked-in columns
        int   i  = i0 + tid;
        int   mi = mask[b * A_ + i];
        sum += pb3[0] * (float)(nm - mi);               // hoisted pair_b3 bias
        util[b * A_ + i] = mi ? (base_u[b * A_ + i] + sum) : -1.0e9f;
    }
}

// ---------------------------------------------------------------------------
// K3: masked row softmax over A per batch.
// ---------------------------------------------------------------------------
__global__ void __launch_bounds__(128)
softmax_kernel(const float* __restrict__ util, float* __restrict__ out)
{
    __shared__ float red[A_];
    const int b = blockIdx.x, t = threadIdx.x;
    float u = util[b * A_ + t];
    red[t] = u; __syncthreads();
    for (int s = 64; s > 0; s >>= 1) {
        if (t < s) red[t] = fmaxf(red[t], red[t + s]);
        __syncthreads();
    }
    float mx = red[0]; __syncthreads();
    float e = __expf(u - mx);
    red[t] = e; __syncthreads();
    for (int s = 64; s > 0; s >>= 1) {
        if (t < s) red[t] = red[t] + red[t + s];
        __syncthreads();
    }
    out[b * A_ + t] = e / red[0];
}

// ---------------------------------------------------------------------------
extern "C" void kernel_launch(void* const* d_in, const int* in_sizes, int n_in,
                              void* d_out, int out_size, void* d_ws, size_t ws_size,
                              hipStream_t stream) {
    const float* X    = (const float*)d_in[0];
    const int*   mask = (const int*)  d_in[1];
    const float* bW1  = (const float*)d_in[2];
    const float* bb1  = (const float*)d_in[3];
    const float* bW2  = (const float*)d_in[4];
    const float* bb2  = (const float*)d_in[5];
    const float* bW3  = (const float*)d_in[6];
    const float* bb3  = (const float*)d_in[7];
    const float* pW1  = (const float*)d_in[8];
    const float* pb1  = (const float*)d_in[9];
    const float* pW2  = (const float*)d_in[10];
    const float* pb2  = (const float*)d_in[11];
    const float* pW3  = (const float*)d_in[12];
    const float* pb3  = (const float*)d_in[13];

    float* base_u = (float*)d_ws;            // B_*A_ floats
    float* util   = base_u + B_ * A_;        // B_*A_ floats
    float* out    = (float*)d_out;

    base_kernel<<<(B_ * A_) / 256, 256, 0, stream>>>(X, bW1, bb1, bW2, bb2, bW3, bb3, base_u);
    pair_kernel<<<dim3(A_ / 16, B_), 128, 0, stream>>>(X, mask, pW1, pb1, pW2, pb2, pW3, pb3,
                                                       base_u, util);
    softmax_kernel<<<B_, 128, 0, stream>>>(util, out);
}
